// Mamba2LikeBlock_68917045232169
// MI455X (gfx1250) — matrix operand order repaired and checked by reading, not compiled, not run on hardware
//
#include <hip/hip_runtime.h>
#include <hip/hip_bf16.h>

// ---------------------------------------------------------------------------
// Mamba2-like block for MI455X (gfx1250, wave32, WMMA).
//   K0: f32 -> bf16 streaming convert (u, Win, Wout)        [bandwidth-bound]
//   K1: proj = u @ Win^T    (bf16 WMMA, 4 M-tiles/wave, b128 fragment loads)
//   K2: layernorm(Bp/Cp), DT=softplus(dd_dt+bias), dec=exp(A*DT)
//   K3: selective scan, 1 WG per (batch,head), 64x64 state in registers,
//       per-step operands staged via global_load_async_to_lds (ASYNCcnt)
//   K4: out = Y @ Wout^T    (bf16 WMMA)
// ---------------------------------------------------------------------------

typedef __attribute__((ext_vector_type(16))) __bf16 v16bf;
typedef __attribute__((ext_vector_type(8)))  __bf16 v8bf;
typedef __attribute__((ext_vector_type(8)))  float  v8f;

static __device__ __forceinline__ unsigned short bfbits(float f) {
    unsigned u = __builtin_bit_cast(unsigned, f);
    return (unsigned short)((u + 0x7FFFu + ((u >> 16) & 1u)) >> 16);
}
static __device__ __forceinline__ __bf16 f2bf(float f) {
    return __builtin_bit_cast(__bf16, bfbits(f));
}
static __device__ __forceinline__ float softplusf(float x) {
    return (x > 20.f) ? x : log1pf(expf(x));
}

// ---- K0: streaming f32 -> bf16 (RNE), 4 elements / thread --------------------
__global__ void __launch_bounds__(256)
cvt_bf16_kernel(const float* __restrict__ src, __bf16* __restrict__ dst, int n4) {
    const int i = blockIdx.x * 256 + threadIdx.x;
    if (i < n4) {
        const float4 v = ((const float4*)src)[i];
        ushort4 o;
        o.x = bfbits(v.x); o.y = bfbits(v.y); o.z = bfbits(v.z); o.w = bfbits(v.w);
        ((ushort4*)dst)[i] = o;
    }
}

// ---- WMMA GEMM:  C[m,n] = sum_k A[m,k]*Bw[n,k],  A:MxK, Bw:NxK (bf16) -------
// One wave per 64x16 output (4 M-tiles, shared B fragment), K stepped by 32.
__global__ void __launch_bounds__(32)
wmma_gemm_nt_bf16(const __bf16* __restrict__ A, const __bf16* __restrict__ Bw,
                  float* __restrict__ C, int M, int N, int K) {
    const int lane = threadIdx.x;
    const int H    = lane >> 4;          // half-wave select
    const int mn   = lane & 15;
    const int tm0  = blockIdx.y << 6;    // 64-row macro tile
    const int tn   = blockIdx.x << 4;

    const __bf16* brow = Bw + (size_t)(tn + mn) * K + (H << 4);
    const __bf16* arow0 = A + (size_t)(tm0 + mn) * K + (H << 4);
    const size_t  astep = (size_t)16 * K;   // 16 rows

    v8f c0 = {}, c1 = {}, c2 = {}, c3 = {};
    for (int k0 = 0; k0 < K; k0 += 32) {
        __builtin_prefetch(brow + k0 + 64, 0, 0);    // global_prefetch_b8
        __builtin_prefetch(arow0 + k0 + 64, 0, 0);
        // B 32x16 fragment: lane = N, elements e -> K = k0 + 16H + e (contiguous)
        const v8bf blo = *(const v8bf*)(brow + k0);
        const v8bf bhi = *(const v8bf*)(brow + k0 + 8);
        const v16bf b = __builtin_shufflevector(blo, bhi,
                            0,1,2,3,4,5,6,7,8,9,10,11,12,13,14,15);
        // A 16x32 fragment: lane = M, elements e -> K = k0 + 16H + e (contiguous)
        v16bf a[4];
#pragma unroll
        for (int mi = 0; mi < 4; ++mi) {
            const __bf16* ar = arow0 + mi * astep + k0;
            const v8bf alo = *(const v8bf*)(ar);
            const v8bf ahi = *(const v8bf*)(ar + 8);
            a[mi] = __builtin_shufflevector(alo, ahi,
                        0,1,2,3,4,5,6,7,8,9,10,11,12,13,14,15);
        }
        c0 = __builtin_amdgcn_wmma_f32_16x16x32_bf16(false, a[0], false, b, (short)0, c0, false, false);
        c1 = __builtin_amdgcn_wmma_f32_16x16x32_bf16(false, a[1], false, b, (short)0, c1, false, false);
        c2 = __builtin_amdgcn_wmma_f32_16x16x32_bf16(false, a[2], false, b, (short)0, c2, false, false);
        c3 = __builtin_amdgcn_wmma_f32_16x16x32_bf16(false, a[3], false, b, (short)0, c3, false, false);
    }
    // C/D layout: VGPR v -> M = v + 8H, lane = N
    v8f cc[4] = {c0, c1, c2, c3};
#pragma unroll
    for (int mi = 0; mi < 4; ++mi)
#pragma unroll
        for (int v = 0; v < 8; ++v)
            C[(size_t)(tm0 + mi * 16 + v + 8 * H) * N + tn + mn] = cc[mi][v];
}

// ---- layernorm over 64 elements with one wave (2 elems/lane) ----------------
static __device__ __forceinline__ void ln64(const float* __restrict__ src,
                                            const float* __restrict__ g,
                                            const float* __restrict__ bta,
                                            float* __restrict__ dst, int lane) {
    float x0 = src[lane], x1 = src[lane + 32];
    float s1 = x0 + x1, s2 = x0 * x0 + x1 * x1;
#pragma unroll
    for (int m = 16; m >= 1; m >>= 1) {
        s1 += __shfl_xor(s1, m, 32);
        s2 += __shfl_xor(s2, m, 32);
    }
    const float mean = s1 * (1.f / 64.f);
    const float var  = s2 * (1.f / 64.f) - mean * mean;
    const float rstd = rsqrtf(var + 1e-5f);
    dst[lane]      = (x0 - mean) * rstd * g[lane]      + bta[lane];
    dst[lane + 32] = (x1 - mean) * rstd * g[lane + 32] + bta[lane + 32];
}

// ---- K2: per-token prep -----------------------------------------------------
__global__ void __launch_bounds__(32)
prep_kernel(const float* __restrict__ proj, const float* __restrict__ dt_bias,
            const float* __restrict__ Bg, const float* __restrict__ Bb,
            const float* __restrict__ Cg, const float* __restrict__ Cb,
            float* __restrict__ Bn, float* __restrict__ Cn,
            float* __restrict__ DT, float* __restrict__ DEC) {
    const int t    = blockIdx.x;
    const int lane = threadIdx.x;
    const float* p = proj + (size_t)t * 3248;

    ln64(p + 3072, Bg, Bb, Bn + (size_t)t * 64, lane);
    ln64(p + 3136, Cg, Cb, Cn + (size_t)t * 64, lane);

    if (lane < 24) {
        const float dt = softplusf(p[3200 + lane] + dt_bias[lane]);
        const float a  = fminf(-softplusf(p[3224 + lane]), -1e-4f);
        DT [(size_t)t * 24 + lane] = dt;
        DEC[(size_t)t * 24 + lane] = expf(a * dt);
    }
}

// ---- async global -> LDS (ASYNCcnt path, CDNA5 ISA) ------
static __device__ __forceinline__ void async_g2lds_b32(float* lds_dst,
                                                       const float* gsrc) {
    const unsigned lds_off = (unsigned)(unsigned long long)(void*)lds_dst;
    asm volatile("global_load_async_to_lds_b32 %0, %1, off"
                 :: "v"(lds_off), "v"(gsrc) : "memory");
}
static __device__ __forceinline__ void wait_async0() {
    asm volatile("s_wait_asynccnt 0x0" ::: "memory");
}

// ---- K3: selective scan, one 256-thread WG per (b,h) ------------------------
// thread t owns state rows p = t>>2, columns n in [(t&3)*16, +16)
__global__ void __launch_bounds__(256)
scan_kernel(const float* __restrict__ proj, const float* __restrict__ Bn,
            const float* __restrict__ Cn, const float* __restrict__ DT,
            const float* __restrict__ DEC, const float* __restrict__ Dp,
            __bf16* __restrict__ Yb) {
    const int bh  = blockIdx.x;
    const int b   = bh / 24, h = bh % 24;
    const int tid = threadIdx.x;
    __shared__ float xs[64], Bs[64], Cs[64], sc[2];

    float hreg[16];
#pragma unroll
    for (int i = 0; i < 16; ++i) hreg[i] = 0.f;

    const int   p  = tid >> 2;
    const int   n0 = (tid & 3) << 4;
    const float Dv = Dp[h];

    for (int t = 0; t < 1024; ++t) {
        const size_t tok = (size_t)b * 1024 + t;
        __syncthreads();   // previous-step LDS reads complete before overwrite
        if (tid < 64)        async_g2lds_b32(&xs[tid],       proj + tok * 3248 + 1536 + h * 64 + tid);
        else if (tid < 128)  async_g2lds_b32(&Bs[tid - 64],  Bn + tok * 64 + (tid - 64));
        else if (tid < 192)  async_g2lds_b32(&Cs[tid - 128], Cn + tok * 64 + (tid - 128));
        else if (tid == 192) { sc[0] = DT[tok * 24 + h]; sc[1] = DEC[tok * 24 + h]; }
        wait_async0();
        __syncthreads();

        const float xp = xs[p], dt = sc[0], dc = sc[1];
        const float c0 = dt * xp;
        float acc = 0.f;
#pragma unroll
        for (int i = 0; i < 16; ++i) {
            const int n = n0 + i;
            hreg[i] = dc * hreg[i] + c0 * Bs[n];     // h = dec*h + dt*outer(x,Bn)
            acc    += hreg[i] * Cs[n];               // y = h @ Cn
        }
        acc += __shfl_xor(acc, 1, 32);               // 4 lanes per p, same wave
        acc += __shfl_xor(acc, 2, 32);
        if ((tid & 3) == 0) {
            const float z   = proj[tok * 3248 + h * 64 + p];
            const float sil = z / (1.f + expf(-z));
            Yb[tok * 1536 + h * 64 + p] = f2bf((acc + Dv * xp) * sil);
        }
    }
}

extern "C" void kernel_launch(void* const* d_in, const int* in_sizes, int n_in,
                              void* d_out, int out_size, void* d_ws, size_t ws_size,
                              hipStream_t stream) {
    (void)in_sizes; (void)n_in; (void)out_size; (void)ws_size;
    const float* u       = (const float*)d_in[0];   // (2,1024,768)
    const float* Win     = (const float*)d_in[1];   // (3248,768)
    const float* Wout    = (const float*)d_in[2];   // (768,1536)
    const float* dt_bias = (const float*)d_in[3];
    const float* Dp      = (const float*)d_in[4];
    const float* Bg      = (const float*)d_in[5];
    const float* Bb      = (const float*)d_in[6];
    const float* Cg      = (const float*)d_in[7];
    const float* Cb      = (const float*)d_in[8];
    float* out = (float*)d_out;                     // (2,1024,768)

    // workspace layout: f32 region first, then bf16 region (all 16B aligned)
    float* proj = (float*)d_ws;                        // 2048*3248 f32
    float* BnW  = proj + (size_t)2048 * 3248;          // 2048*64
    float* CnW  = BnW  + (size_t)2048 * 64;
    float* DTW  = CnW  + (size_t)2048 * 64;            // 2048*24
    float* DEC  = DTW  + (size_t)2048 * 24;
    __bf16* ub    = (__bf16*)(DEC + (size_t)2048 * 24); // 2048*768  bf16
    __bf16* Winb  = ub    + (size_t)2048 * 768;         // 3248*768
    __bf16* Woutb = Winb  + (size_t)3248 * 768;         // 768*1536
    __bf16* Yb    = Woutb + (size_t)768 * 1536;         // 2048*1536

    // K0: convert inputs to bf16 (streaming, bandwidth-bound)
    cvt_bf16_kernel<<<(2048 * 768 / 4 + 255) / 256, 256, 0, stream>>>(u, ub, 2048 * 768 / 4);
    cvt_bf16_kernel<<<(3248 * 768 / 4 + 255) / 256, 256, 0, stream>>>(Win, Winb, 3248 * 768 / 4);
    cvt_bf16_kernel<<<(768 * 1536 / 4 + 255) / 256, 256, 0, stream>>>(Wout, Woutb, 768 * 1536 / 4);

    // K1: proj = u @ Win^T   (M=2048, N=3248, K=768)
    wmma_gemm_nt_bf16<<<dim3(3248 / 16, 2048 / 64), 32, 0, stream>>>(ub, Winb, proj,
                                                                     2048, 3248, 768);
    // K2: per-token prep
    prep_kernel<<<2048, 32, 0, stream>>>(proj, dt_bias, Bg, Bb, Cg, Cb,
                                         BnW, CnW, DTW, DEC);
    // K3: scan over L, 48 (b,h) workgroups; writes bf16 Y directly
    scan_kernel<<<48, 256, 0, stream>>>(proj, BnW, CnW, DTW, DEC, Dp, Yb);
    // K4: out = Y @ Wout^T   (M=2048, N=768, K=1536)
    wmma_gemm_nt_bf16<<<dim3(768 / 16, 2048 / 64), 32, 0, stream>>>(Yb, Woutb, out,
                                                                    2048, 768, 1536);
}
